// RNN_46660524703965
// MI455X (gfx1250) — compile-verified, matrix operand order checked by reading, not run on hardware
//
#include <hip/hip_runtime.h>

typedef __attribute__((ext_vector_type(16))) _Float16 v16h;
typedef __attribute__((ext_vector_type(8)))  float    v8f;
typedef __attribute__((ext_vector_type(2)))  float    v2f;

#define T_STEPS 2048
#define BATCH   2048
#define HID     32

#if __has_builtin(__builtin_amdgcn_tanhf)
// Native V_TANH_F32 on gfx1250
__device__ __forceinline__ float fast_tanh(float x) {
    return __builtin_amdgcn_tanhf(x);
}
// sigmoid(x) where v == x/2 already (weights pre-scaled by 0.5): 1 trans + 1 fma
__device__ __forceinline__ float sig_from_half(float v) {
    return fmaf(__builtin_amdgcn_tanhf(v), 0.5f, 0.5f);
}
#else
__device__ __forceinline__ float fast_tanh(float x) {
    return __builtin_amdgcn_rcpf(1.0f + __expf(-2.0f * x)) * 2.0f - 1.0f;
}
// sigmoid(x) with v == x/2: 1/(1+e^-2v)
__device__ __forceinline__ float sig_from_half(float v) {
    return __builtin_amdgcn_rcpf(1.0f + __expf(-2.0f * v));
}
#endif

// One wave handles a 16-wide batch tile for all T steps.
// gates^T = W_hh(A, f16 16x32 tiles) x h^T(B, f16 32x16)   [8 gate tiles]
//         + W_ih*x + bias via V_WMMA_F32_16X16X4_F32 (A2=[Wih|bias|0|0], B2=[x;1;0;0])
// Sigmoid-gate tiles (i,f,o = tiles 0,1,2,3,6,7) have weights pre-scaled by 0.5
// so sigmoid = fma(v_tanh(acc), 0.5, 0.5) with no per-step pre-multiply.
__global__ __launch_bounds__(128, 1)
void lstm_wmma_kernel(const float* __restrict__ x,
                      const float* __restrict__ W_ih,
                      const float* __restrict__ W_hh,
                      const float* __restrict__ b_ih,
                      const float* __restrict__ b_hh,
                      const float* __restrict__ W_out,
                      const float* __restrict__ b_out,
                      float* __restrict__ out)
{
    const int  lane   = threadIdx.x & 31;
    const int  lanelo = lane & 15;
    const bool hi     = lane >= 16;                       // upper half-wave
    const int  wave   = (blockIdx.x * blockDim.x + threadIdx.x) >> 5;
    const int  b      = wave * 16 + lanelo;               // this lane's batch column

    // ---- W_hh tiles in f16 A-matrix layout (16x32 per tile, 8 tiles for 4H=128 rows)
    // lane<16 : M=lanelo, K = {0..7, 16..23}; lane>=16: M=lanelo, K = {8..15, 24..31}
    const int koff = hi ? 8 : 0;
    v16h Ahh[8];
    #pragma unroll
    for (int t = 0; t < 8; ++t) {
        const float scale = (t == 4 || t == 5) ? 1.0f : 0.5f;  // g-gate unscaled
        const float* row = W_hh + (t * 16 + lanelo) * HID;
        #pragma unroll
        for (int e = 0; e < 8; ++e) {
            Ahh[t][e]     = (_Float16)(scale * row[koff + e]);
            Ahh[t][8 + e] = (_Float16)(scale * row[16 + koff + e]);
        }
    }

    // ---- A2 tiles (f32 16x4): col0 = W_ih, col1 = b_ih+b_hh, cols 2,3 = 0
    // f32 A layout: lanes<16 hold K=0,1 in V0,V1; lanes>=16 hold K=2,3 (zeros here)
    v2f Axb[8];
    #pragma unroll
    for (int t = 0; t < 8; ++t) {
        const float scale = (t == 4 || t == 5) ? 1.0f : 0.5f;
        const int g = t * 16 + lanelo;
        Axb[t].x = hi ? 0.0f : (scale * W_ih[g]);
        Axb[t].y = hi ? 0.0f : (scale * (b_ih[g] + b_hh[g]));
    }

    // ---- per-lane W_out coefficients for this lane's 16 (batch,unit) slots
    float wo[16];
    #pragma unroll
    for (int k = 0; k < 16; ++k) {
        const int u = ((k < 8) ? 0 : 16) + (hi ? 8 : 0) + (k & 7);
        wo[k] = W_out[u];
    }
    const float bo  = b_out[0];
    const float byc = hi ? 0.0f : 1.0f;   // constant row-1 of B2 (hoisted)

    // ---- recurrent state: h (f16, B-matrix layout) and c (f32), fully in registers
    v16h hB = {};
    float cst[16], hs[16];
    #pragma unroll
    for (int k = 0; k < 16; ++k) { cst[k] = 0.0f; hs[k] = 0.0f; }

    // ---- 2-deep branch-free x prefetch pipeline (load at t is consumed at t+2)
    float xcur = x[b];                       // x[0, b]
    float xnxt = x[(size_t)BATCH + b];       // x[1, b]   (T >= 2)

    for (int t = 0; t < T_STEPS; ++t) {
        // long-range L2/L1 prefetch: no LOADcnt, keeps the t+2 load cache-hot
        const int tp = (t + 16 < T_STEPS) ? (t + 16) : (T_STEPS - 1);
        __builtin_prefetch(&x[(size_t)tp * BATCH + b], 0, 3);

        const int t2 = (t + 2 < T_STEPS) ? (t + 2) : (T_STEPS - 1);
        const float xfut = x[(size_t)t2 * BATCH + b];   // unconditional, clamped

        // B2 (f32 4x16): row0 = x, row1 = 1 -> lanes<16 carry K0,K1; upper lanes zero
        v2f Bx;
        Bx.x = hi ? 0.0f : xcur;
        Bx.y = byc;

        // gates^T accumulation: acc[g] covers gate rows [16g, 16g+16)
        v8f acc[8];
        #pragma unroll
        for (int g = 0; g < 8; ++g) {
            v8f z = {};
            acc[g] = __builtin_amdgcn_wmma_f32_16x16x4_f32(
                false, Axb[g], false, Bx, (short)0, z, false, false);
            acc[g] = __builtin_amdgcn_wmma_f32_16x16x32_f16(
                false, Ahh[g], false, hB, (short)0, acc[g], false, false);
        }

        // per-lane LSTM pointwise update (i,f,g,o live at matching slots)
        float partial = 0.0f;
        #pragma unroll
        for (int k = 0; k < 16; ++k) {
            const int s   = k & 7;
            const int off = (k < 8) ? 0 : 1;
            const float iv = sig_from_half(acc[0 + off][s]);   // pre-scaled
            const float fv = sig_from_half(acc[2 + off][s]);   // pre-scaled
            const float gv = fast_tanh    (acc[4 + off][s]);   // unscaled
            const float ov = sig_from_half(acc[6 + off][s]);   // pre-scaled
            const float cn = fv * cst[k] + iv * gv;
            cst[k] = cn;
            const float hv = ov * fast_tanh(cn);
            hs[k] = hv;
            partial += hv * wo[k];
        }
        // repack h into f16 B-matrix layout for the next step (pure per-lane)
        #pragma unroll
        for (int k = 0; k < 16; ++k) hB[k] = (_Float16)hs[k];

        // O==1 output: combine the two half-wave partial dots, lanes<16 store
        const float full = partial + __shfl_xor(partial, 16, 32);
        if (!hi) out[(size_t)t * BATCH + b] = full + bo;

        xcur = xnxt;
        xnxt = xfut;
    }

    // ---- final states: d_out = [outs (T*B)] [hT (B*H)] [cT (B*H)]
    float* hT = out + (size_t)T_STEPS * BATCH;
    float* cT = hT + (size_t)BATCH * HID;
    #pragma unroll
    for (int k = 0; k < 16; ++k) {
        const int u = ((k < 8) ? 0 : 16) + (hi ? 8 : 0) + (k & 7);
        hT[(size_t)b * HID + u] = hs[k];
        cT[(size_t)b * HID + u] = cst[k];
    }
}

extern "C" void kernel_launch(void* const* d_in, const int* in_sizes, int n_in,
                              void* d_out, int out_size, void* d_ws, size_t ws_size,
                              hipStream_t stream) {
    const float* x     = (const float*)d_in[0];
    const float* W_ih  = (const float*)d_in[1];
    const float* W_hh  = (const float*)d_in[2];
    const float* b_ih  = (const float*)d_in[3];
    const float* b_hh  = (const float*)d_in[4];
    const float* W_out = (const float*)d_in[5];
    const float* b_out = (const float*)d_in[6];
    float* out = (float*)d_out;

    // 2048 batch / 16 per wave = 128 waves = 32 blocks x 4 waves
    dim3 grid(BATCH / 16 / 4), block(128);
    hipLaunchKernelGGL(lstm_wmma_kernel, grid, block, 0, stream,
                       x, W_ih, W_hh, b_ih, b_hh, W_out, b_out, out);
}
